// LocalGaussianBlur_72988674228770
// MI455X (gfx1250) — matrix operand order repaired and checked by reading, not compiled, not run on hardware
//
#include <hip/hip_runtime.h>
#include <stdint.h>

typedef __attribute__((ext_vector_type(16))) _Float16 v16h;
typedef __attribute__((ext_vector_type(8)))  float    v8f;

#define IMG_H 512
#define IMG_W 512
#define KS    11
#define PAD   5
#define TILE_W 16          // pixels per wave (one row)
#define TILE_H 8           // rows per block (one per wave, 8 waves = 256 threads)
#define LW 27              // LDS row stride (26 valid cols + 1 pad, odd to spread banks)
#define LH (TILE_H + KS - 1)   // 18
#define LDS_ELEMS (3 * LH * LW)

__global__ __launch_bounds__(256)
void lgb_wmma_kernel(const float* __restrict__ img,
                     const float* __restrict__ mod,
                     float* __restrict__ out)
{
    __shared__ float tile[3][LH][LW];

    const int tid = threadIdx.x;
    const int x0  = blockIdx.x * TILE_W;
    const int y0  = blockIdx.y * TILE_H;

    // ---------- stage haloed image tile into LDS via async-to-LDS DMA ----------
    for (int e = tid; e < LDS_ELEMS; e += 256) {
        int c  = e / (LH * LW);
        int r  = e - c * (LH * LW);
        int ty = r / LW;
        int tx = r - ty * LW;
        if (tx < TILE_W + KS - 1) {              // 26 valid columns
            int gy = y0 - PAD + ty; gy = gy < 0 ? 0 : (gy > IMG_H - 1 ? IMG_H - 1 : gy);
            int gx = x0 - PAD + tx; gx = gx < 0 ? 0 : (gx > IMG_W - 1 ? IMG_W - 1 : gx);
            uint64_t ga = (uint64_t)(uintptr_t)(img + ((size_t)c * IMG_H + gy) * IMG_W + gx);
            uint32_t la = (uint32_t)(uintptr_t)(&tile[c][ty][tx]);   // low 32 bits = LDS offset
            asm volatile("global_load_async_to_lds_b32 %0, %1, off"
                         :: "v"(la), "v"(ga) : "memory");
        }
    }
    asm volatile("s_wait_asynccnt 0" ::: "memory");
    __syncthreads();

    // ---------- per-pixel Gaussian weights (lane pair L, L+16 both serve pixel L&15) ----------
    const int  lane = tid & 31;
    const int  wv   = tid >> 5;        // wave id = row in tile
    const int  pix  = lane & 15;       // which of the 16 pixels this lane serves
    const bool hiL  = lane >= 16;
    const int  y = y0 + wv;
    const int  x = x0 + pix;

    float sd   = mod[(size_t)y * IMG_W + x];
    float rvar = 1.0f / (2.0f * sd * sd + 1e-8f);
    float w1[KS];
    float s = 0.0f;
#pragma unroll
    for (int i = 0; i < KS; ++i) {
        float n = (float)(i - PAD);
        w1[i] = __expf(-(n * n) * rvar);     // v_exp_f32
        s += w1[i];
    }
    const float inv = 1.0f / (s * s);        // k2d normalizer = (sum k1d)^2

    const float* base0 = &tile[0][wv][pix];
    const float* base1 = &tile[1][wv][pix];
    const float* base2 = &tile[2][wv][pix];

    v8f C0 = {}; v8f C1 = {}; v8f C2 = {};

#pragma unroll
    for (int chunk = 0; chunk < 4; ++chunk) {
        // ----- B fragment: per-pixel 2D weights, split f16 hi/lo -----
        // B 32x16 layout: lanes 0-15 K = chunk*32 + t + (t>=8?8:0); lanes 16-31: +8
        v16h Whi, Wlo;
#pragma unroll
        for (int t = 0; t < 16; ++t) {
            const int kA = chunk * 32 + t + ((t >= 8) ? 8 : 0);       // compile-time -> w1[] indices constant
            const int kB = kA + 8;
            float wa = (kA < 121) ? w1[kA / 11] * w1[kA % 11] : 0.0f;
            float wb = (kB < 121) ? w1[kB / 11] * w1[kB % 11] : 0.0f;
            float wsel = hiL ? wb : wa;
            _Float16 h = (_Float16)wsel;
            Whi[t] = h;
            Wlo[t] = (_Float16)(wsel - (float)h);
        }

        // ----- A fragment per channel + 3 compensated WMMAs -----
        // A 16x32 layout: lanes 0-15 K = chunk*32 + t + (t>=8?8:0); lanes 16-31: +8
        auto do_channel = [&](const float* bp, v8f c) -> v8f {
            v16h Ahi, Alo;
#pragma unroll
            for (int t = 0; t < 16; ++t) {
                const int kA = chunk * 32 + t + ((t >= 8) ? 8 : 0);   // <= 119, always valid
                const int kB = kA + 8;
                float va = bp[(kA / 11) * LW + (kA % 11)];            // ds_load const offset
                float vb = (kB < 121) ? bp[(kB / 11) * LW + (kB % 11)] : 0.0f;
                float vsel = hiL ? vb : va;
                _Float16 h = (_Float16)vsel;
                Ahi[t] = h;
                Alo[t] = (_Float16)(vsel - (float)h);
            }
            // error-compensated product: A*W = Ahi*Wlo + Alo*Whi + Ahi*Whi (drop lo*lo)
            c = __builtin_amdgcn_wmma_f32_16x16x32_f16(false, Ahi, false, Wlo, (short)0, c, false, false);
            c = __builtin_amdgcn_wmma_f32_16x16x32_f16(false, Alo, false, Whi, (short)0, c, false, false);
            c = __builtin_amdgcn_wmma_f32_16x16x32_f16(false, Ahi, false, Whi, (short)0, c, false, false);
            return c;
        };
        C0 = do_channel(base0, C0);
        C1 = do_channel(base1, C1);
        C2 = do_channel(base2, C2);
    }

    // ---------- extract diagonal D[i][i] and store ----------
    // i<8: lane i, VGPR i ; i>=8: lane i+16, VGPR i-8.  Each lane shares its element sel=lane&7.
    const int sel     = lane & 7;
    const int srcLane = (pix < 8) ? pix : (pix + 16);
#pragma unroll
    for (int ch = 0; ch < 3; ++ch) {
        v8f c = (ch == 0) ? C0 : ((ch == 1) ? C1 : C2);
        float v = c[0];
#pragma unroll
        for (int r = 1; r < 8; ++r) v = (sel == r) ? c[r] : v;
        float dv = __shfl(v, srcLane, 32);
        if (lane < 16)
            out[((size_t)ch * IMG_H + y) * IMG_W + x] = dv * inv;
    }
}

extern "C" void kernel_launch(void* const* d_in, const int* in_sizes, int n_in,
                              void* d_out, int out_size, void* d_ws, size_t ws_size,
                              hipStream_t stream)
{
    (void)in_sizes; (void)n_in; (void)out_size; (void)d_ws; (void)ws_size;
    const float* img = (const float*)d_in[0];   // (1,3,512,512) f32
    const float* mod = (const float*)d_in[1];   // (512,512) f32
    float* out = (float*)d_out;                 // (1,3,512,512) f32

    dim3 grid(IMG_W / TILE_W, IMG_H / TILE_H);  // 32 x 64 = 2048 blocks
    lgb_wmma_kernel<<<grid, 256, 0, stream>>>(img, mod, out);
}
